// VGGReplacementDistillGraspNet_16733192585547
// MI455X (gfx1250) — matrix operand order repaired and checked by reading, not compiled, not run on hardware
//
#include <hip/hip_runtime.h>
#include <hip/hip_bf16.h>

// ---------------------------------------------------------------------------
// Problem constants (from reference setup_inputs)
// ---------------------------------------------------------------------------
#define BB   8
#define NS   2048
#define KK   16384
#define DD   768
#define CC   256
#define NCOL (BB * NS)          // 16384 flattened columns
#define LN_EPS 1e-5f

typedef __bf16 bf16x16 __attribute__((ext_vector_type(16)));
typedef float  f32x8   __attribute__((ext_vector_type(8)));

union Frag32B { bf16x16 v; uint4 u[2]; };

__device__ __forceinline__ unsigned short f32_to_bf16(float f) {
    unsigned int x = __float_as_uint(f);
    x += 0x7FFFu + ((x >> 16) & 1u);          // round-to-nearest-even
    return (unsigned short)(x >> 16);
}

// ---------------------------------------------------------------------------
// f32 -> bf16 weight conversion
// ---------------------------------------------------------------------------
__global__ __launch_bounds__(256) void cvt_bf16_kernel(const float* __restrict__ s,
                                                       unsigned short* __restrict__ d,
                                                       int n) {
    int i = blockIdx.x * 256 + threadIdx.x;
    if (i < n) d[i] = f32_to_bf16(s[i]);
}

// ---------------------------------------------------------------------------
// 1-NN argmin: for each seed find nearest of K pts (raw coords — the
// per-batch centroid/scale normalization cancels inside the argmin).
// grid (NS/256, B), block 256. pts tiles streamed memory->LDS with CDNA5
// async loads (ASYNCcnt), bypassing VGPRs in this VALU-bound kernel.
// ---------------------------------------------------------------------------
__global__ __launch_bounds__(256) void nn_argmin_kernel(const float* __restrict__ seed,
                                                        const float* __restrict__ pts,
                                                        int* __restrict__ idx) {
    const int b   = blockIdx.y;
    const int tid = threadIdx.x;
    const int n   = blockIdx.x * 256 + tid;
    const float* s = seed + ((size_t)b * NS + n) * 3;
    const float sx = s[0], sy = s[1], sz = s[2];
    const float* pbase = pts + (size_t)b * KK * 3;

    __shared__ float sp[512 * 3] __attribute__((aligned(16)));
    // LDS byte offset = low 32 bits of the flat shared address
    const unsigned lds_base = (unsigned)(size_t)(void*)sp;

    float best = 3.4e38f;
    int   bi   = 0;
    for (int t0 = 0; t0 < KK; t0 += 512) {
        __syncthreads();
        {   // async tile copy: 6144 B = 256 lanes x 16 B + 128 lanes x 16 B
            unsigned           l0 = lds_base + tid * 16;
            unsigned long long g0 =
                (unsigned long long)((const char*)pbase + (size_t)t0 * 12 + tid * 16);
            asm volatile("global_load_async_to_lds_b128 %0, %1, off"
                         :: "v"(l0), "v"(g0) : "memory");
            if (tid < 128) {
                unsigned           l1 = l0 + 4096;
                unsigned long long g1 = g0 + 4096;
                asm volatile("global_load_async_to_lds_b128 %0, %1, off"
                             :: "v"(l1), "v"(g1) : "memory");
            }
            // speculative prefetch of the next tile -> global_prefetch_b8
            if (t0 + 512 < KK)
                __builtin_prefetch((const char*)pbase + (size_t)(t0 + 512) * 12 + tid * 24, 0, 0);
            asm volatile("s_wait_asynccnt 0x0" ::: "memory");
        }
        __syncthreads();
#pragma unroll 4
        for (int j = 0; j < 512; ++j) {
            float dx = sx - sp[j * 3 + 0];
            float dy = sy - sp[j * 3 + 1];
            float dz = sz - sp[j * 3 + 2];
            float dd = dx * dx + dy * dy + dz * dz;
            if (dd < best) { best = dd; bi = t0 + j; }   // strict '<': first-min
        }
    }
    idx[b * NS + n] = bi;
}

// ---------------------------------------------------------------------------
// Gather: G[col][d] (bf16, K-major for GEMM B-fragments) = feat[b][d][idx]
// grid NCOL, block 256 (3 channels per thread).
// ---------------------------------------------------------------------------
__global__ __launch_bounds__(256) void gather_feat_kernel(const float* __restrict__ feat,
                                                          const int* __restrict__ idx,
                                                          unsigned short* __restrict__ G) {
    const int col = blockIdx.x;
    const int b   = col >> 11;                 // /2048
    const int k   = idx[col];
    const float* src = feat + (size_t)b * DD * KK + k;
    unsigned short* dst = G + (size_t)col * DD;
    for (int d = threadIdx.x; d < DD; d += 256)
        dst[d] = f32_to_bf16(src[(size_t)d * KK]);
}

__global__ void zero_loss_kernel(float* p) { *p = 0.0f; }

// ---------------------------------------------------------------------------
// WMMA GEMM: Out[256 x 64cols] per block, 8 waves, wave -> 32 rows x 64 cols
// = 2 Mtiles x 4 Ntiles of v_wmma_f32_16x16x32_bf16.
// A = Wbf [256][KDIM] (L2-resident), B = Bmat [col][KDIM] (streamed once).
// EPI: 0 = bias + LayerNorm(channel) -> bf16 [col][256]
//      1 = bias + ReLU              -> bf16 [col][256]
//      2 = bias -> f32 d_out[b][o][n] + MSE-loss atomic accumulate
// ---------------------------------------------------------------------------
template <int EPI>
__global__ __launch_bounds__(256)
void gemm_wmma_kernel(const unsigned short* __restrict__ Wbf,
                      const unsigned short* __restrict__ Bmat,
                      const float* __restrict__ bias,
                      int KDIM,
                      unsigned short* __restrict__ OutBf,
                      float* __restrict__ OutF,
                      const float* __restrict__ fteacher,
                      float* __restrict__ lossAcc,
                      float invTotal) {
    const int tid  = threadIdx.x;
    const int wave = tid >> 5;
    const int lane = tid & 31;
    const int l16  = lane & 15;
    const int h    = lane >> 4;
    const int colg0 = blockIdx.x * 64;
    const int mbase = wave * 32;

    f32x8 acc[2][4];
#pragma unroll
    for (int mt = 0; mt < 2; ++mt)
#pragma unroll
        for (int nt = 0; nt < 4; ++nt) acc[mt][nt] = (f32x8)0.0f;

    for (int k0 = 0; k0 < KDIM; k0 += 32) {
        Frag32B a[2], bfr[4];
        // A fragments: lane holds row m, k-runs [k0+8h,+8) and [k0+16+8h,+8)
#pragma unroll
        for (int mt = 0; mt < 2; ++mt) {
            const unsigned short* ap = Wbf + (size_t)(mbase + mt * 16 + l16) * KDIM + k0;
            a[mt].u[0] = *(const uint4*)(ap + h * 8);
            a[mt].u[1] = *(const uint4*)(ap + 16 + h * 8);
        }
        // B fragments: lane holds col n, k-runs [k0+8h,+8) and [k0+16+8h,+8)
#pragma unroll
        for (int nt = 0; nt < 4; ++nt) {
            const unsigned short* bp =
                Bmat + (size_t)(colg0 + nt * 16 + l16) * KDIM + k0 + h * 8;
            bfr[nt].u[0] = *(const uint4*)(bp);
            bfr[nt].u[1] = *(const uint4*)(bp + 16);
        }
#pragma unroll
        for (int mt = 0; mt < 2; ++mt)
#pragma unroll
            for (int nt = 0; nt < 4; ++nt)
                acc[mt][nt] = __builtin_amdgcn_wmma_f32_16x16x32_bf16(
                    false, a[mt].v, false, bfr[nt].v, (short)0, acc[mt][nt],
                    false, false);
    }

    // bias (acc element r, lane: row = mbase + mt*16 + 8h + r, col = nt*16+l16)
#pragma unroll
    for (int mt = 0; mt < 2; ++mt)
#pragma unroll
        for (int r = 0; r < 8; ++r) {
            float bv = bias[mbase + mt * 16 + 8 * h + r];
#pragma unroll
            for (int nt = 0; nt < 4; ++nt) acc[mt][nt][r] += bv;
        }

    if constexpr (EPI == 0) {
        __shared__ float red[128];                 // [64 cols][sum, sumsq]
        __shared__ unsigned short tbuf[64 * 264];  // padded transpose buffer
        if (tid < 128) red[tid] = 0.0f;
        __syncthreads();
#pragma unroll
        for (int nt = 0; nt < 4; ++nt) {
            float s = 0.f, q = 0.f;
#pragma unroll
            for (int mt = 0; mt < 2; ++mt)
#pragma unroll
                for (int r = 0; r < 8; ++r) {
                    float v = acc[mt][nt][r];
                    s += v; q += v * v;
                }
            int n = nt * 16 + l16;
            atomicAdd(&red[2 * n], s);       // ds_add_f32
            atomicAdd(&red[2 * n + 1], q);
        }
        __syncthreads();
        if (tid < 64) {
            float mu  = red[2 * tid] * (1.0f / 256.0f);
            float var = red[2 * tid + 1] * (1.0f / 256.0f) - mu * mu;
            red[2 * tid]     = mu;
            red[2 * tid + 1] = __frsqrt_rn(var + LN_EPS);
        }
        __syncthreads();
#pragma unroll
        for (int nt = 0; nt < 4; ++nt) {
            int n = nt * 16 + l16;
            float mu = red[2 * n], rs = red[2 * n + 1];
#pragma unroll
            for (int mt = 0; mt < 2; ++mt)
#pragma unroll
                for (int r = 0; r < 8; ++r) {
                    int m = mbase + mt * 16 + 8 * h + r;
                    tbuf[n * 264 + m] = f32_to_bf16((acc[mt][nt][r] - mu) * rs);
                }
        }
        __syncthreads();
        {   // coalesced bf16 store: OutBf[col][256]
            int n = tid >> 2, c0 = (tid & 3) * 64;
            unsigned short* op = OutBf + (size_t)(colg0 + n) * 256 + c0;
#pragma unroll
            for (int j = 0; j < 8; ++j)
                *(uint4*)(op + j * 8) = *(const uint4*)(&tbuf[n * 264 + c0 + j * 8]);
        }
    } else if constexpr (EPI == 1) {
        __shared__ unsigned short tbuf[64 * 264];
#pragma unroll
        for (int nt = 0; nt < 4; ++nt) {
            int n = nt * 16 + l16;
#pragma unroll
            for (int mt = 0; mt < 2; ++mt)
#pragma unroll
                for (int r = 0; r < 8; ++r) {
                    int m = mbase + mt * 16 + 8 * h + r;
                    float v = acc[mt][nt][r];
                    tbuf[n * 264 + m] = f32_to_bf16(v > 0.f ? v : 0.f);
                }
        }
        __syncthreads();
        int n = tid >> 2, c0 = (tid & 3) * 64;
        unsigned short* op = OutBf + (size_t)(colg0 + n) * 256 + c0;
#pragma unroll
        for (int j = 0; j < 8; ++j)
            *(uint4*)(op + j * 8) = *(const uint4*)(&tbuf[n * 264 + c0 + j * 8]);
    } else {
        __shared__ float obuf[256 * 65];           // padded [row][col] f32
#pragma unroll
        for (int nt = 0; nt < 4; ++nt) {
            int n = nt * 16 + l16;
#pragma unroll
            for (int mt = 0; mt < 2; ++mt)
#pragma unroll
                for (int r = 0; r < 8; ++r) {
                    int m = mbase + mt * 16 + 8 * h + r;
                    obuf[m * 65 + n] = acc[mt][nt][r];
                }
        }
        __syncthreads();
        const int o  = tid;
        const int b  = colg0 >> 11;
        const int n0 = colg0 & 2047;
        const float* ft = fteacher + ((size_t)b * CC + o) * NS + n0;
        float*       fo = OutF     + ((size_t)b * CC + o) * NS + n0;
        float ls = 0.f;
#pragma unroll 4
        for (int j = 0; j < 64; ++j) {
            float v = obuf[o * 65 + j];
            float d = v - ft[j];
            ls += d * d;
            fo[j] = v;
        }
#pragma unroll
        for (int off = 16; off > 0; off >>= 1)
            ls += __shfl_down(ls, off, 32);
        if ((tid & 31) == 0) atomicAdd(lossAcc, ls * invTotal);
    }
}

// ---------------------------------------------------------------------------
// Host launcher
// ---------------------------------------------------------------------------
extern "C" void kernel_launch(void* const* d_in, const int* in_sizes, int n_in,
                              void* d_out, int out_size, void* d_ws, size_t ws_size,
                              hipStream_t stream) {
    (void)in_sizes; (void)n_in; (void)ws_size;
    // setup_inputs order
    const float* seed_xyz  = (const float*)d_in[1];
    const float* pts       = (const float*)d_in[2];
    const float* feat      = (const float*)d_in[3];
    const float* f_teacher = (const float*)d_in[4];
    const float* W_proj    = (const float*)d_in[5];
    const float* b_proj    = (const float*)d_in[6];
    const float* W_a1      = (const float*)d_in[7];
    const float* b_a1      = (const float*)d_in[8];
    const float* W_a2      = (const float*)d_in[9];
    const float* b_a2      = (const float*)d_in[10];

    // carve workspace
    char* ws = (char*)d_ws;
    size_t off = 0;
    auto carve = [&](size_t bytes) {
        char* p = ws + off;
        off = (off + bytes + 255) & ~(size_t)255;
        return p;
    };
    int*            idx = (int*)           carve((size_t)NCOL * 4);
    unsigned short* Wp  = (unsigned short*)carve((size_t)CC * DD * 2);
    unsigned short* Wa1 = (unsigned short*)carve((size_t)CC * CC * 2);
    unsigned short* Wa2 = (unsigned short*)carve((size_t)CC * CC * 2);
    unsigned short* G   = (unsigned short*)carve((size_t)NCOL * DD * 2);
    unsigned short* Xln = (unsigned short*)carve((size_t)NCOL * CC * 2);
    unsigned short* H   = (unsigned short*)carve((size_t)NCOL * CC * 2);

    float* outF    = (float*)d_out;
    float* lossPtr = outF + (out_size - 1);

    // 1) weights -> bf16
    cvt_bf16_kernel<<<(CC * DD + 255) / 256, 256, 0, stream>>>(W_proj, Wp, CC * DD);
    cvt_bf16_kernel<<<(CC * CC + 255) / 256, 256, 0, stream>>>(W_a1, Wa1, CC * CC);
    cvt_bf16_kernel<<<(CC * CC + 255) / 256, 256, 0, stream>>>(W_a2, Wa2, CC * CC);

    // 2) 1-NN argmin (normalization cancels in argmin)
    nn_argmin_kernel<<<dim3(NS / 256, BB), 256, 0, stream>>>(seed_xyz, pts, idx);

    // 3) gather to bf16 K-major matrix
    gather_feat_kernel<<<NCOL, 256, 0, stream>>>(feat, idx, G);

    // 4) zero loss accumulator (graph-replay safe)
    zero_loss_kernel<<<1, 1, 0, stream>>>(lossPtr);

    // 5) proj GEMM + LayerNorm
    gemm_wmma_kernel<0><<<NCOL / 64, 256, 0, stream>>>(
        Wp, G, b_proj, DD, Xln, nullptr, nullptr, nullptr, 0.f);
    // 6) adapter1 GEMM + ReLU
    gemm_wmma_kernel<1><<<NCOL / 64, 256, 0, stream>>>(
        Wa1, Xln, b_a1, CC, H, nullptr, nullptr, nullptr, 0.f);
    // 7) adapter2 GEMM + output + loss
    const float invTotal = 1.0f / ((float)BB * CC * NS);
    gemm_wmma_kernel<2><<<NCOL / 64, 256, 0, stream>>>(
        Wa2, H, b_a2, CC, nullptr, outF, f_teacher, lossPtr, invTotal);
}
